// GeneralisedShapeletTransform_22789096472760
// MI455X (gfx1250) — matrix operand order repaired and checked by reading, not compiled, not run on hardware
//
#include <hip/hip_runtime.h>
#include <math.h>

// Problem constants from the reference: T=128, B=8, C=6, M=32, K=16, S=64.
#define T_N 128
#define B_N 8
#define C_N 6
#define M_N 32
#define K_N 16
#define S_N 64
#define NODES (T_N + K_N)          // 144 merged knots
#define NSEG  (NODES - 1)          // 143 segments
#define JTOT  (NSEG * C_N)         // 858 (segment, channel) feature columns
#define NCHUNK ((JTOT + 31) / 32)  // 27 WMMA K-chunks of 32
#define PATH_ELEMS (B_N * T_N * C_N)  // 6144 floats = 24 KB

typedef __attribute__((ext_vector_type(16))) _Float16 v16h;
typedef __attribute__((ext_vector_type(8)))  float    v8f;
typedef __attribute__((ext_vector_type(4)))  unsigned int u32x4;
typedef __attribute__((ext_vector_type(4)))  int          i32x4;
typedef __attribute__((ext_vector_type(8)))  int          i32x8;

#if __has_builtin(__builtin_amdgcn_tensor_load_to_lds) && \
    __has_builtin(__builtin_amdgcn_s_wait_tensorcnt)
#define USE_TDM 1
#if __has_include(<hip/amd_detail/amd_gfx1250_TDM.h>)
#define TDM_ARGS6 1   // amdgpu-toolchain: 6-arg builtin
#endif
#endif

__global__ void gst_init_out(float* out) {
    int i = blockIdx.x * blockDim.x + threadIdx.x;
    if (i < B_N * M_N) ((unsigned int*)out)[i] = 0x7F800000u; // +inf
}

// One wave32 workgroup per (m, s) candidate. All WMMA calls are in uniform
// control flow so EXEC is all-ones (ISA requirement).
__launch_bounds__(32)
__global__ void gst_kernel(const float* __restrict__ path,       // (B,T,C)
                           const float* __restrict__ lengths,    // (M)
                           const float* __restrict__ shapelets,  // (M,K,C)
                           float* __restrict__ out) {            // (B,M)
    __shared__ float    ldsPath[PATH_ELEMS];       // 24 KB: whole path
    __shared__ float    ldsShp[K_N * C_N];         // this shapelet
    __shared__ float    ldsKnots[K_N];
    __shared__ float    ldsTs[NODES];              // merged sorted knot times
    __shared__ _Float16 Xh[16 * 32];               // one 16x32 f16 chunk of X
    __shared__ float    diag[32 * 8];              // accumulator spill for diag

    const int lane = threadIdx.x;
    const int m = blockIdx.x / S_N;
    const int s = blockIdx.x % S_N;

    const float len   = lengths[m];
    const float start = (float)s * (1.0f / (float)(S_N - 1)) * (127.0f - len);
    const float end   = start + len;
    const float inv_h = (float)(K_N - 1) / len;    // 1/h, h = len/(K-1)

    // ---- stage path into LDS ----
#ifdef USE_TDM
    // Tensor Data Mover: one async DMA of the contiguous 24 KB path block.
    // D# built per CDNA5 ISA (2D descriptor, groups 2/3 zero).
    {
        unsigned long long ga = (unsigned long long)path;          // 57-bit VA
        unsigned int lds_off  = (unsigned int)(unsigned long long)ldsPath; // addr[31:0] = LDS byte offset
        u32x4 g0 = {};
        g0[0] = 1u;                                    // count=1, user mode
        g0[1] = lds_off;                               // D#.lds_addr
        g0[2] = (unsigned int)(ga & 0xFFFFFFFFu);      // global_addr[31:0]
        g0[3] = (unsigned int)((ga >> 32) & 0x1FFFFFFu) | (2u << 30); // [56:32] | type=2
        i32x8 g1 = {};
        g1[0] = 0x00020000;                            // wg_mask=0, data_size=2 (4B)
        g1[1] = (int)((PATH_ELEMS & 0xFFFF) << 16);    // tensor_dim0[15:0] (bits 63:48)
        g1[2] = (int)(((PATH_ELEMS >> 16) & 0xFFFF) | (1u << 16)); // dim0[31:16], tensor_dim1=1
        g1[3] = (int)((PATH_ELEMS & 0xFFFF) << 16);    // tile_dim0 = 6144 (bits 127:112)
        g1[4] = 1;                                     // tile_dim1=1, tile_dim2=0
        g1[5] = (int)PATH_ELEMS;                       // tensor_dim0_stride[31:0]
        g1[6] = (int)((PATH_ELEMS & 0xFFFF) << 16);    // stride0[47:32]=0, stride1[15:0]
        g1[7] = 0;                                     // stride1[47:16]=0
        i32x4 g2 = {};
        i32x4 g3 = {};
#ifdef TDM_ARGS6
        i32x8 g4 = {};
        __builtin_amdgcn_tensor_load_to_lds(g0, g1, g2, g3, g4, 0);
#else
        __builtin_amdgcn_tensor_load_to_lds(g0, g1, g2, g3, 0);
#endif
    }
#else
    {
        const float4* g4p = (const float4*)path;
        float4*       l4  = (float4*)ldsPath;
        for (int i = lane; i < PATH_ELEMS / 4; i += 32) l4[i] = g4p[i];
    }
#endif

    // ---- stage shapelet slice + knots (overlaps with the TDM transfer) ----
    for (int i = lane; i < K_N * C_N; i += 32)
        ldsShp[i] = shapelets[m * K_N * C_N + i];
    if (lane < K_N)
        ldsKnots[lane] = start + len * ((float)lane * (1.0f / (float)(K_N - 1)));
    __syncthreads();

    // ---- build merged-sorted ts by rank scatter (== stable sort of concat) ----
    if (lane < K_N) {                               // knot rank = k + #times<=g
        float g = ldsKnots[lane];
        int cnt = (int)floorf(g) + 1;               // times are 0..127
        cnt = min(max(cnt, 0), T_N);
        ldsTs[lane + cnt] = g;
    }
    for (int i = lane; i < T_N; i += 32) {          // time rank = i + #knots<i
        float tv = (float)i;
        int cnt = 0;
        #pragma unroll
        for (int k = 0; k < K_N; ++k) cnt += (ldsKnots[k] < tv) ? 1 : 0;
        ldsTs[i + cnt] = tv;
    }
    __syncthreads();
    for (int i = lane; i < NODES; i += 32)          // clip (order-preserving)
        ldsTs[i] = fminf(fmaxf(ldsTs[i], start), end);

#ifdef USE_TDM
    __builtin_amdgcn_s_wait_tensorcnt(0);           // path DMA complete
#endif
    __syncthreads();

    // ---- Gram-diagonal via WMMA: disc^2[b] = sum_j X[b,j]^2 + X[b+8,j]^2 ----
    // X rows 0..7 : sqrt(dt/4) *(d_n + d_{n+1})   (per channel)
    // X rows 8..15: sqrt(dt/12)*(d_n - d_{n+1})
    v8f acc = {};
    for (int chk = 0; chk < NCHUNK; ++chk) {
        const int j = chk * 32 + lane;              // this lane's (n,c) column
        float col[16];
        #pragma unroll
        for (int r = 0; r < 16; ++r) col[r] = 0.0f;
        if (j < JTOT) {
            const int n = j / C_N;
            const int c = j - n * C_N;
            const float x0 = ldsTs[n], x1 = ldsTs[n + 1];
            const float dt = x1 - x0;
            const float wsum = sqrtf(dt * 0.25f);
            const float wdif = sqrtf(dt * (1.0f / 12.0f));
            // shapelet interpolation at x0,x1 (channel c)
            float rel0 = (x0 - start) * inv_h;
            int   si0  = min(max((int)floorf(rel0), 0), K_N - 2);
            float sw0  = rel0 - (float)si0;
            float q0 = ldsShp[si0 * C_N + c] * (1.0f - sw0) +
                       ldsShp[(si0 + 1) * C_N + c] * sw0;
            float rel1 = (x1 - start) * inv_h;
            int   si1  = min(max((int)floorf(rel1), 0), K_N - 2);
            float sw1  = rel1 - (float)si1;
            float q1 = ldsShp[si1 * C_N + c] * (1.0f - sw1) +
                       ldsShp[(si1 + 1) * C_N + c] * sw1;
            // path interpolation (unit-spaced times => w = x - idx)
            int   i0 = min(max((int)floorf(x0), 0), T_N - 2);
            float f0 = x0 - (float)i0;
            int   i1 = min(max((int)floorf(x1), 0), T_N - 2);
            float f1 = x1 - (float)i1;
            #pragma unroll
            for (int b = 0; b < B_N; ++b) {
                const float* pb = &ldsPath[(b * T_N) * C_N];
                float p0 = pb[i0 * C_N + c] * (1.0f - f0) + pb[(i0 + 1) * C_N + c] * f0;
                float p1 = pb[i1 * C_N + c] * (1.0f - f1) + pb[(i1 + 1) * C_N + c] * f1;
                float d0 = p0 - q0;
                float d1 = p1 - q1;
                col[b]     = wsum * (d0 + d1);
                col[8 + b] = wdif * (d0 - d1);
            }
        }
        #pragma unroll
        for (int r = 0; r < 16; ++r) Xh[r * 32 + lane] = (_Float16)col[r];
        __syncthreads();

        // Fragments per ISA 7.12.2 (wave32, 16-bit):
        //   A 16x32: lane L holds row M=L%16; K = (e&7) + 16*(e>>3) + 8*(L>=16)
        //   B 32x16 (= X^T): lane L holds col N=L%16; K = (e&7) + 16*(e>>3) + 8*(L>=16)
        v16h afrag, bfrag;
        const int mrow  = lane & 15;
        const int ahalf = (lane >> 4) << 3;   // 0 or 8
        const int bbase = (lane >> 4) << 4;   // 0 or 16
        #pragma unroll
        for (int e = 0; e < 16; ++e) {
            int ka = (e & 7) + ((e >> 3) << 4) + ahalf;
            afrag[e] = Xh[mrow * 32 + ka];
        }
        #pragma unroll
        for (int e = 0; e < 16; ++e)
            bfrag[e] = Xh[mrow * 32 + (e & 7) + ((e >> 3) << 4) + ahalf];   // B[k,n] = X[n,k]

        acc = __builtin_amdgcn_wmma_f32_16x16x32_f16(
            /*neg_a=*/false, afrag, /*neg_b=*/false, bfrag,
            /*c_mod=*/(short)0, acc, /*reuse_a=*/false, /*reuse_b=*/false);
        __syncthreads();   // protect Xh before next chunk overwrites it
    }

    // ---- extract Gram diagonal (C layout: VGPR r, lane l -> M=r+8*(l>=16), N=l%16)
    #pragma unroll
    for (int r = 0; r < 8; ++r) diag[lane * 8 + r] = acc[r];
    __syncthreads();
    if (lane < B_N) {
        // (b,b) lives in lane b, VGPR b ; (b+8,b+8) lives in lane b+24, VGPR b
        float d2   = diag[lane * 8 + lane] + diag[(lane + 24) * 8 + lane];
        float disc = sqrtf(fmaxf(d2, 0.0f));
        // min over S via uint atomicMin (valid: disc >= 0)
        atomicMin((unsigned int*)out + (lane * M_N + m), __float_as_uint(disc));
    }
}

extern "C" void kernel_launch(void* const* d_in, const int* in_sizes, int n_in,
                              void* d_out, int out_size, void* d_ws, size_t ws_size,
                              hipStream_t stream) {
    // inputs: 0=times(128, unused: arange), 1=path(8*128*6), 2=lengths(32),
    //         3=shapelets(32*16*6); output: (8,32) f32
    const float* path      = (const float*)d_in[1];
    const float* lengths   = (const float*)d_in[2];
    const float* shapelets = (const float*)d_in[3];
    float* out = (float*)d_out;

    gst_init_out<<<1, 256, 0, stream>>>(out);
    gst_kernel<<<M_N * S_N, 32, 0, stream>>>(path, lengths, shapelets, out);
}
